// ToReference_86766929314313
// MI455X (gfx1250) — compile-verified, matrix operand order checked by reading, not run on hardware
//
#include <hip/hip_runtime.h>

// ---------------- problem constants ----------------
#define KNSUB   16
#define KP      144          // points per subdomain
#define KR      2304         // 48*48
#define KS      12           // SAMPLE_SIZE
#define KHID    256
#define KCOUT   32
#define KBATCH  2
#define KINCH   19
#define KRCHUNK 128
#define KNCHUNK (KR / KRCHUNK)       // 18

// ---------------- LDS layout (dynamic shared) ----------------
#define LDS_HTILE   0                                   // 144*256 f32  = 147456 B
#define LDS_GSUM    (KP * KHID * 4)                     // 128*256 bf16 =  65536 B
#define LDS_W2      (LDS_GSUM + KRCHUNK * KHID * 2)     //  32*256 bf16 =  16384 B
#define LDS_W1REF   (LDS_W2 + KCOUT * KHID * 2)         // 256 f32      =   1024 B
#define LDS_SIDX    (LDS_W1REF + KHID * 4)              // 128*12 i32   =   6144 B
#define LDS_B2      (LDS_SIDX + KRCHUNK * KS * 4)       //  32 f32      =    128 B
#define LDS_BYTES   (LDS_B2 + KCOUT * 4)                // 236672 B  (<= 320KB WGP LDS)

typedef __attribute__((ext_vector_type(16))) __bf16        v16bf;
typedef __attribute__((ext_vector_type(8)))  float         v8f;
typedef __attribute__((ext_vector_type(4)))  unsigned int  v4u;
typedef __attribute__((ext_vector_type(8)))  int           v8i;
typedef __attribute__((ext_vector_type(4)))  int           v4i;

__device__ __forceinline__ unsigned rotl32(unsigned x, int n) {
  return (x << n) | (x >> (32 - n));
}

// Exact JAX threefry2x32 for key(42); returns raw 32-bit draw for flat index i
// of a (16,2304,12) randint tensor (split-halves counter layout).
__device__ unsigned threefry42(unsigned i) {
  const unsigned HALF = (KNSUB * KR * KS) / 2u;   // 221184
  unsigned x0, x1;
  const bool hi = (i >= HALF);
  if (hi) { x0 = i - HALF; x1 = i; } else { x0 = i; x1 = i + HALF; }
  const unsigned ks0 = 0u, ks1 = 42u;
  const unsigned ks2 = 0x1BD11BDAu ^ ks0 ^ ks1;
  x0 += ks0; x1 += ks1;
#define TF_R(r) { x0 += x1; x1 = rotl32(x1, (r)); x1 ^= x0; }
  TF_R(13) TF_R(15) TF_R(26) TF_R(6)   x0 += ks1; x1 += ks2 + 1u;
  TF_R(17) TF_R(29) TF_R(16) TF_R(24)  x0 += ks2; x1 += ks0 + 2u;
  TF_R(13) TF_R(15) TF_R(26) TF_R(6)   x0 += ks0; x1 += ks1 + 3u;
  TF_R(17) TF_R(29) TF_R(16) TF_R(24)  x0 += ks1; x1 += ks2 + 4u;
  TF_R(13) TF_R(15) TF_R(26) TF_R(6)   x0 += ks2; x1 += ks0 + 5u;
#undef TF_R
  return hi ? x1 : x0;
}

// tanh-approx GELU (jax.nn.gelu default), exp via v_exp_f32 + v_rcp_f32
__device__ __forceinline__ float gelu_tanh(float x) {
  float u = 0.7978845608028654f * (x + 0.044715f * x * x * x);
  float e = __expf(2.0f * u);
  float t = 1.0f - 2.0f * __builtin_amdgcn_rcpf(e + 1.0f);
  return 0.5f * x * (1.0f + t);
}

__device__ __forceinline__ unsigned short f2bf(float f) {
  unsigned u = __float_as_uint(f);
  unsigned r = u + 0x7FFFu + ((u >> 16) & 1u);
  return (unsigned short)(r >> 16);
}

// ---------------- kernel 0: stable argsort(subdomain_lookup) -> order[16][144]
__global__ void build_order_kernel(const long long* __restrict__ lookup,
                                   int* __restrict__ order) {
  int n = threadIdx.x;
  if (n >= KNSUB) return;
  int c = 0;
  for (int i = 0; i < KR; ++i) {
    if ((int)lookup[i] == n) order[n * KP + c++] = i;   // stable: ascending i
  }
}

// ---------------- kernel 1: h_partial[b][n][p][h] = b1 + W1[:,0:2]*(px,py) + W1[:,3:19]*v
// (ref-coordinate channel W1[:,2] applied later, per r)
__global__ void hpartial_kernel(const float* __restrict__ v,
                                const float* __restrict__ pc,
                                const int* __restrict__ order,
                                const float* __restrict__ W1,
                                const float* __restrict__ b1,
                                float* __restrict__ hp) {
  const int h  = threadIdx.x;          // 256 threads = hidden dim
  const int bn = blockIdx.x;           // 32 blocks = (b,n)
  const int b  = bn / KNSUB;
  const int n  = bn % KNSUB;

  float w[KINCH];
#pragma unroll
  for (int c = 0; c < KINCH; ++c) w[c] = W1[h * KINCH + c];
  const float bias = b1[h];

  for (int p = 0; p < KP; ++p) {
    const int g = order[n * KP + p];
    float acc = bias + w[0] * pc[g] + w[1] * pc[KR + g];
#pragma unroll
    for (int c = 0; c < 16; ++c)
      acc += w[3 + c] * v[((size_t)(b * 16 + c)) * KR + g];
    hp[(((size_t)bn * KP) + p) * KHID + h] = acc;
  }
}

// ---------------- kernel 2: TDM-staged gather + GELU-sum + bf16 WMMA layer-2
__global__ void __launch_bounds__(256)
fused_mlp_kernel(const float* __restrict__ hp,
                 const float* __restrict__ W1,
                 const float* __restrict__ W2,
                 const float* __restrict__ b2,
                 float* __restrict__ out) {
  extern __shared__ char smem[];
  float*          h_tile = (float*)(smem + LDS_HTILE);
  unsigned short* gsum   = (unsigned short*)(smem + LDS_GSUM);
  unsigned short* w2s    = (unsigned short*)(smem + LDS_W2);
  float*          w1ref  = (float*)(smem + LDS_W1REF);
  int*            sidx   = (int*)(smem + LDS_SIDX);
  float*          b2s    = (float*)(smem + LDS_B2);

  const int tid  = threadIdx.x;
  const int lane = tid & 31;
  const int wv   = __builtin_amdgcn_readfirstlane(tid >> 5);   // wave id 0..7
  const int bid  = blockIdx.x;
  const int chunk = bid % KNCHUNK;
  const int n     = (bid / KNCHUNK) % KNSUB;
  const int b     = bid / (KNCHUNK * KNSUB);
  const int r0    = chunk * KRCHUNK;

  // --- TDM: each wave DMAs 18 rows (18*256 f32 = 18432 B) of h_partial -> LDS
  {
    const unsigned lds0 = __builtin_amdgcn_groupstaticsize();  // dynamic-LDS base
    unsigned long long ga = (unsigned long long)(const void*)
        (hp + ((size_t)(b * KNSUB + n) * KP + (size_t)wv * 18) * KHID);
    v4u g0;
    g0[0] = 1u;                                            // count = 1 (valid D#)
    g0[1] = lds0 + (unsigned)(wv * 18432);                 // lds_addr
    g0[2] = (unsigned)(ga & 0xFFFFFFFFull);                // global_addr[31:0]
    g0[3] = (unsigned)((ga >> 32) & 0x1FFFFFFull)          // global_addr[56:32]
            | (2u << 30);                                  // type = 2 ("image")
    v8i g1;
    g1[0] = (int)(3u << 16);       // workgroup_mask=0, data_size=3 (8 bytes)
    g1[1] = (int)(2304u << 16);    // tensor_dim0[15:0] = 2304 (8B units)
    g1[2] = (int)(1u << 16);       // tensor_dim0 hi = 0 | tensor_dim1 = 1
    g1[3] = (int)(2304u << 16);    // tile_dim0 = 2304
    g1[4] = 0;                     // tile_dim1/2 unused
    g1[5] = 2304;                  // tensor_dim0_stride low32
    g1[6] = 0;
    g1[7] = 0;
    v4i z4 = {0, 0, 0, 0};
    v8i z8 = {0, 0, 0, 0, 0, 0, 0, 0};
    __builtin_amdgcn_tensor_load_to_lds(g0, g1, z4, z4, z8, 0);
  }

  // --- stage small operands + Threefry sample indices while the TDM flies
  for (int i = tid; i < KCOUT * KHID; i += 256) w2s[i] = f2bf(W2[i]);
  if (tid < KHID)  w1ref[tid] = W1[tid * KINCH + 2];
  if (tid < KCOUT) b2s[tid]   = b2[tid];
  for (int j = tid; j < KRCHUNK * KS; j += 256) {
    int rr = j / KS, s = j - rr * KS;
    unsigned i = (unsigned)((n * KR + (r0 + rr)) * KS + s);
    sidx[j] = (int)(threefry42(i) % 143u);                 // randint(0, P-1)
  }

  __builtin_amdgcn_s_wait_tensorcnt(0);
  __syncthreads();

  // --- phase B: g_sum[rr][h] = sum_s gelu(h_partial[idx] + w1ref[h]*ref(r))
  // Lane owns the contiguous hidden slice h = 8*lane .. 8*lane+7 so every
  // gather is two ds_load_b128 and the bf16 result is one ds_store_b128.
  const int h0 = lane * 8;
  float wr[8];
  {
    const float4 wa = *(const float4*)(w1ref + h0);
    const float4 wb = *(const float4*)(w1ref + h0 + 4);
    wr[0] = wa.x; wr[1] = wa.y; wr[2] = wa.z; wr[3] = wa.w;
    wr[4] = wb.x; wr[5] = wb.y; wr[6] = wb.z; wr[7] = wb.w;
  }
  for (int rr = wv; rr < KRCHUNK; rr += 8) {
    const float refc = -1.0f + 2.0f * (float)(r0 + rr) / (float)(KR - 1);
    float acc[8];
#pragma unroll
    for (int k = 0; k < 8; ++k) acc[k] = 0.0f;
    for (int s = 0; s < KS; ++s) {
      const int idx = sidx[rr * KS + s];
      const float4* row4 = (const float4*)(h_tile + idx * KHID + h0);
      const float4 xa = row4[0];
      const float4 xb = row4[1];
      acc[0] += gelu_tanh(xa.x + wr[0] * refc);
      acc[1] += gelu_tanh(xa.y + wr[1] * refc);
      acc[2] += gelu_tanh(xa.z + wr[2] * refc);
      acc[3] += gelu_tanh(xa.w + wr[3] * refc);
      acc[4] += gelu_tanh(xb.x + wr[4] * refc);
      acc[5] += gelu_tanh(xb.y + wr[5] * refc);
      acc[6] += gelu_tanh(xb.z + wr[6] * refc);
      acc[7] += gelu_tanh(xb.w + wr[7] * refc);
    }
    uint4 pk;
    pk.x = (unsigned)f2bf(acc[0]) | ((unsigned)f2bf(acc[1]) << 16);
    pk.y = (unsigned)f2bf(acc[2]) | ((unsigned)f2bf(acc[3]) << 16);
    pk.z = (unsigned)f2bf(acc[4]) | ((unsigned)f2bf(acc[5]) << 16);
    pk.w = (unsigned)f2bf(acc[6]) | ((unsigned)f2bf(acc[7]) << 16);
    *(uint4*)(gsum + rr * KHID + h0) = pk;
  }
  __syncthreads();

  // --- phase C: D[128 x 32] = g_sum[128 x 256] * W2^T[256 x 32] via WMMA bf16
  // 16 output tiles (8 M-tiles x 2 N-tiles); 2 tiles per wave; K loop = 8.
  union Frag { v16bf v; unsigned u[8]; };
  const int hi = lane >> 4;        // half-wave select (K +8 offset)
  const int lm = lane & 15;
  for (int t = wv; t < 16; t += 8) {
    const int mt = t & 7;
    const int nt = t >> 3;
    v8f c = {};
    const unsigned* arow = (const unsigned*)(gsum + (mt * 16 + lm) * KHID);
    const unsigned* brow = (const unsigned*)(w2s  + (nt * 16 + lm) * KHID);
#pragma unroll
    for (int kt = 0; kt < 8; ++kt) {
      Frag a, bf;
#pragma unroll
      for (int j = 0; j < 8; ++j) {
        // element pair (2j,2j+1) holds K = 2j + 8*((j>=4) + (lane>=16))
        const int K = 2 * j + ((j >= 4) ? 8 : 0) + hi * 8;
        a.u[j]  = arow[kt * 16 + (K >> 1)];
        bf.u[j] = brow[kt * 16 + (K >> 1)];
      }
      c = __builtin_amdgcn_wmma_f32_16x16x32_bf16(
              false, a.v, false, bf.v, (short)0, c, false, false);
    }
    // D layout: lane lm = col N, VGPR v = row M = v + 8*hi
    const int o = nt * 16 + lm;
    const float bias = b2s[o];
    const size_t base = ((size_t)((b * KNSUB + n) * KCOUT + o)) * KR
                        + (size_t)(r0 + mt * 16);
#pragma unroll
    for (int vreg = 0; vreg < 8; ++vreg) {
      const int M = vreg + 8 * hi;
      out[base + M] = c[vreg] * (1.0f / 12.0f) + bias;
    }
  }
}

extern "C" void kernel_launch(void* const* d_in, const int* in_sizes, int n_in,
                              void* d_out, int out_size, void* d_ws, size_t ws_size,
                              hipStream_t stream) {
  (void)in_sizes; (void)n_in; (void)out_size; (void)ws_size;
  const float*     v      = (const float*)d_in[0];
  const float*     pc     = (const float*)d_in[1];
  const long long* lookup = (const long long*)d_in[2];
  const float*     W1     = (const float*)d_in[3];
  const float*     b1     = (const float*)d_in[4];
  const float*     W2     = (const float*)d_in[5];
  const float*     b2     = (const float*)d_in[6];
  float*           out    = (float*)d_out;

  int*   order = (int*)d_ws;                           //  9216 B
  float* hp    = (float*)((char*)d_ws + 16384);        //  2*16*144*256 f32 = 4.7 MB

  build_order_kernel<<<1, 16, 0, stream>>>(lookup, order);
  hpartial_kernel<<<KBATCH * KNSUB, KHID, 0, stream>>>(v, pc, order, W1, b1, hp);
  fused_mlp_kernel<<<KBATCH * KNSUB * KNCHUNK, 256, LDS_BYTES, stream>>>(
      hp, W1, W2, b2, out);
}
